// OutputLayer_8701603741845
// MI455X (gfx1250) — matrix operand order repaired and checked by reading, not compiled, not run on hardware
//
#include <hip/hip_runtime.h>

#define BB 8
#define LL 2500
#define DD 512
#define YY 8921

typedef __attribute__((ext_vector_type(16))) __bf16 v16bf;
typedef __attribute__((ext_vector_type(8)))  __bf16 v8bf;
typedef __attribute__((ext_vector_type(8)))  float  v8f;

#define XT_STR 24                         // bf16 per xT row: 16 data + 8 pad (48B)
#define TILE_ELEMS (DD * XT_STR)          // 12288 bf16 = 24576 B per 16-col tile
#define TILE_BYTES (TILE_ELEMS * 2)
#define CW 32                             // L-chunk width (2 tiles)
#define NC32 ((LL + CW - 1) / CW)         // 79 chunks
#define NT16 (2 * NC32)                   // 158 16-col tiles (last zero-padded)
#define NTILES ((YY + 15) / 16)           // 558 label tiles
#define TPB 256                           // 8 waves per block
#define WPB 8
#define PRE_BYTES ((size_t)BB * NT16 * TILE_BYTES)   // ~29.7 MB in d_ws

// ---------- CDNA5 async global->LDS copy (ASYNCcnt-tracked) ----------
#if __has_builtin(__builtin_amdgcn_global_load_async_to_lds_b128)
#define HAVE_ASYNC_B128 1
#endif
#if __has_builtin(__builtin_amdgcn_s_wait_asynccnt)
#define HAVE_WAIT_ASYNC 1
#endif

typedef int vi4 __attribute__((__vector_size__(4 * sizeof(int))));
typedef __attribute__((address_space(1))) vi4* gvi4p;   // global int4*
typedef __attribute__((address_space(3))) vi4* lvi4p;   // LDS int4*

__device__ __forceinline__ void async_cp16(const void* g, void* l) {
#if defined(HAVE_ASYNC_B128)
    __builtin_amdgcn_global_load_async_to_lds_b128(
        (gvi4p)(unsigned long long)g,
        (lvi4p)(unsigned)(unsigned long long)l,
        0, 0);
#else
    unsigned lds32 = (unsigned)(unsigned long long)l;
    unsigned long long ga = (unsigned long long)g;
    asm volatile("global_load_async_to_lds_b128 %0, %1, off"
                 :: "v"(lds32), "v"(ga) : "memory");
#endif
}

template <int N>
__device__ __forceinline__ void wait_async_le() {
#if defined(HAVE_WAIT_ASYNC)
    __builtin_amdgcn_s_wait_asynccnt((short)N);
#else
    asm volatile("s_wait_asynccnt %0" :: "i"(N) : "memory");
#endif
}

// ---------- helpers ----------
// A-fragment layout (16x32 bf16): lanes 0-15 row M=lane,   Ks {kk+0..7, kk+16..23}
//                                 lanes 16-31 row M=lane-16, Ks {kk+8..15, kk+24..31}
__device__ __forceinline__ void load_a_frags(const float* __restrict__ W,
                                             int yt0, int lrow, int half,
                                             v16bf* frags) {
    const int yy = yt0 + lrow;
    const bool ok = (yy < YY);
    const float4* P = (const float4*)(W + (size_t)(ok ? yy : 0) * DD);
    #pragma unroll
    for (int f = 0; f < 16; ++f) {
        const int k0 = f * 32 + half * 8;
        float4 q0 = make_float4(0, 0, 0, 0), q1 = q0, q2 = q0, q3 = q0;
        if (ok) {
            q0 = P[k0 / 4];        q1 = P[k0 / 4 + 1];
            q2 = P[(k0 + 16) / 4]; q3 = P[(k0 + 16) / 4 + 1];
        }
        v16bf a;
        a[0]=(__bf16)q0.x;  a[1]=(__bf16)q0.y;  a[2]=(__bf16)q0.z;  a[3]=(__bf16)q0.w;
        a[4]=(__bf16)q1.x;  a[5]=(__bf16)q1.y;  a[6]=(__bf16)q1.z;  a[7]=(__bf16)q1.w;
        a[8]=(__bf16)q2.x;  a[9]=(__bf16)q2.y;  a[10]=(__bf16)q2.z; a[11]=(__bf16)q2.w;
        a[12]=(__bf16)q3.x; a[13]=(__bf16)q3.y; a[14]=(__bf16)q3.z; a[15]=(__bf16)q3.w;
        frags[f] = a;
    }
}

__device__ __forceinline__ v16bf ld_bfrag(const __bf16* base, int row) {
    const int n  = row & 15;
    const int kb = (row & ~31) + ((row >> 4) & 1) * 8;
    v16bf r;
    #pragma unroll
    for (int s = 0; s < 8; ++s) {
        r[s]     = base[(kb + s) * XT_STR + n];
        r[s + 8] = base[(kb + 16 + s) * XT_STR + n];
    }
    return r;
}

__global__ void zero_loss_k(float* p) { *p = 0.0f; }

// ---------- one-shot precompute: x -> bf16, transposed, pre-swizzled tiles ----------
__launch_bounds__(128)
__global__ void xpose_k(const float* __restrict__ x, __bf16* __restrict__ xTg) {
    __shared__ __align__(16) __bf16 tile[TILE_ELEMS];
    const int t = blockIdx.x, b = blockIdx.y, tid = threadIdx.x;
    const int l0 = t * 16;
    const int d  = tid * 4;
    #pragma unroll 4
    for (int it = 0; it < 16; ++it) {
        const int l = l0 + it;
        float4 v = make_float4(0.f, 0.f, 0.f, 0.f);
        if (l < LL) v = ((const float4*)(x + ((size_t)b * LL + l) * DD))[d >> 2];
        tile[(d + 0) * XT_STR + it] = (__bf16)v.x;
        tile[(d + 1) * XT_STR + it] = (__bf16)v.y;
        tile[(d + 2) * XT_STR + it] = (__bf16)v.z;
        tile[(d + 3) * XT_STR + it] = (__bf16)v.w;
    }
    __syncthreads();
    float4* dst = (float4*)(xTg + (size_t)(b * NT16 + t) * TILE_ELEMS);
    const float4* src = (const float4*)tile;
    #pragma unroll
    for (int j = 0; j < 12; ++j) dst[tid + j * 128] = src[tid + j * 128];
}

// ---------- fused main kernel ----------
template <bool PRE>
__launch_bounds__(TPB)
__global__ void laat_fused(const float* __restrict__ x,
                           const __bf16* __restrict__ xTg,
                           const float* __restrict__ target,
                           const float* __restrict__ Uw,
                           const float* __restrict__ Fw,
                           const float* __restrict__ Fb,
                           float* __restrict__ yout,
                           float* __restrict__ loss)
{
    // one 32-col chunk = two 16-col tiles = 48 KB; double-buffered when PRE
    __shared__ __align__(16) __bf16 xT[(PRE ? 2 : 1) * 2 * TILE_ELEMS];

    const int tid  = threadIdx.x;
    const int wave = tid >> 5;
    const int lane = tid & 31;
    const int half = lane >> 4;
    const int lrow = lane & 15;

    const int b   = blockIdx.y;
    const int yt0 = (blockIdx.x * WPB + wave) * 16;

    // U and final_w A-fragments live in VGPRs
    v16bf ufrag[16], wfrag[16];
    load_a_frags(Uw, yt0, lrow, half, ufrag);
    load_a_frags(Fw, yt0, lrow, half, wfrag);

    float rmax[8], rden[8], rnum[8];
    #pragma unroll
    for (int v = 0; v < 8; ++v) { rmax[v] = -1e30f; rden[v] = 0.f; rnum[v] = 0.f; }

    const char* tbase = (const char*)xTg + (size_t)b * NT16 * TILE_BYTES;

    if (PRE) {
        // prologue: async-copy chunk 0 (two tiles, 48 KB linear) into buffer 0
        const char* src = tbase;
        char* dst = (char*)&xT[0];
        #pragma unroll
        for (int j = 0; j < 12; ++j) {
            const int off = (tid + j * TPB) * 16;
            async_cp16(src + off, dst + off);
        }
    }

    for (int c = 0; c < NC32; ++c) {
        const __bf16* xbuf = &xT[(PRE ? (c & 1) : 0) * 2 * TILE_ELEMS];
        const int l0 = c * CW;

        if (PRE) {
            if (c + 1 < NC32) {   // prefetch next chunk into other buffer
                const char* src = tbase + (size_t)(c + 1) * 2 * TILE_BYTES;
                char* dst = (char*)&xT[((c + 1) & 1) * 2 * TILE_ELEMS];
                #pragma unroll
                for (int j = 0; j < 12; ++j) {
                    const int off = (tid + j * TPB) * 16;
                    async_cp16(src + off, dst + off);
                }
                wait_async_le<12>();   // 12 newest outstanding; chunk c landed
            } else {
                wait_async_le<0>();
            }
        } else {
            // in-kernel transpose staging (fallback when d_ws too small)
            __bf16* xw = (__bf16*)xbuf;
            const int dcol = (tid & 127) * 4;
            #pragma unroll 4
            for (int it = 0; it < 16; ++it) {
                const int lr = it * 2 + (tid >> 7);   // 0..31 within chunk
                const int l  = l0 + lr;
                float4 v = make_float4(0.f, 0.f, 0.f, 0.f);
                if (l < LL)
                    v = ((const float4*)(x + ((size_t)b * LL + l) * DD))[dcol >> 2];
                __bf16* tgt = xw + (lr >> 4) * TILE_ELEMS;
                const int col = lr & 15;
                tgt[(dcol + 0) * XT_STR + col] = (__bf16)v.x;
                tgt[(dcol + 1) * XT_STR + col] = (__bf16)v.y;
                tgt[(dcol + 2) * XT_STR + col] = (__bf16)v.z;
                tgt[(dcol + 3) * XT_STR + col] = (__bf16)v.w;
            }
        }
        __syncthreads();   // chunk-c tiles visible to all waves

        // S,T 16x32 result tiles; K = 512 (16 fragments), B-fragments pipelined
        const __bf16* t0 = xbuf;
        const __bf16* t1 = xbuf + TILE_ELEMS;
        v8f sa0 = {0,0,0,0,0,0,0,0}, ta0 = sa0, sa1 = sa0, ta1 = sa0;
        v16bf bf0 = ld_bfrag(t0, lane);
        v16bf bf1 = ld_bfrag(t1, lane);
        #pragma unroll
        for (int f = 0; f < 16; ++f) {
            const v16bf cb0 = bf0;
            const v16bf cb1 = bf1;
            if (f < 15) {   // prefetch next fragment while WMMAs consume current
                bf0 = ld_bfrag(t0, (f + 1) * 32 + lane);
                bf1 = ld_bfrag(t1, (f + 1) * 32 + lane);
            }
            sa0 = __builtin_amdgcn_wmma_f32_16x16x32_bf16(
                      false, ufrag[f], false, cb0, (short)0, sa0, false, false);
            ta0 = __builtin_amdgcn_wmma_f32_16x16x32_bf16(
                      false, wfrag[f], false, cb0, (short)0, ta0, false, false);
            sa1 = __builtin_amdgcn_wmma_f32_16x16x32_bf16(
                      false, ufrag[f], false, cb1, (short)0, sa1, false, false);
            ta1 = __builtin_amdgcn_wmma_f32_16x16x32_bf16(
                      false, wfrag[f], false, cb1, (short)0, ta1, false, false);
        }
        __syncthreads();   // reads done; buffer may be overwritten next iter

        // online softmax-weighted reduction over 32 columns
        const bool ok0 = (l0 + lrow) < LL;
        const bool ok1 = (l0 + 16 + lrow) < LL;
        #pragma unroll
        for (int v = 0; v < 8; ++v) {
            float s0 = ok0 ? sa0[v] : -1e30f;
            float s1 = ok1 ? sa1[v] : -1e30f;
            float tmax = fmaxf(s0, s1);
            tmax = fmaxf(tmax, __shfl_xor(tmax, 1, 32));
            tmax = fmaxf(tmax, __shfl_xor(tmax, 2, 32));
            tmax = fmaxf(tmax, __shfl_xor(tmax, 4, 32));
            tmax = fmaxf(tmax, __shfl_xor(tmax, 8, 32));
            const float nm    = fmaxf(rmax[v], tmax);
            const float scale = __expf(rmax[v] - nm);
            const float p0    = __expf(s0 - nm);
            const float p1    = __expf(s1 - nm);
            const float t0v   = ok0 ? ta0[v] : 0.f;
            const float t1v   = ok1 ? ta1[v] : 0.f;
            float ps = p0 + p1;
            float pt = p0 * t0v + p1 * t1v;
            ps += __shfl_xor(ps, 1, 32); pt += __shfl_xor(pt, 1, 32);
            ps += __shfl_xor(ps, 2, 32); pt += __shfl_xor(pt, 2, 32);
            ps += __shfl_xor(ps, 4, 32); pt += __shfl_xor(pt, 4, 32);
            ps += __shfl_xor(ps, 8, 32); pt += __shfl_xor(pt, 8, 32);
            rden[v] = rden[v] * scale + ps;
            rnum[v] = rnum[v] * scale + pt;
            rmax[v] = nm;
        }
    }

    // finalize: y = num/den + bias, fused mean-BCE partial
    #pragma unroll
    for (int v = 0; v < 8; ++v) {
        const int r  = v + half * 8;
        const int yy = yt0 + r;
        if (lrow == v && yy < YY) {
            const float val = rnum[v] / rden[v] + Fb[yy];
            yout[(size_t)b * YY + yy] = val;
            const float tt = target[(size_t)b * YY + yy];
            const float l1 = fmaxf(val, 0.f) - val * tt +
                             log1pf(__expf(-fabsf(val)));
            atomicAdd(loss, l1 * (1.0f / (BB * (float)YY)));
        }
    }
}

extern "C" void kernel_launch(void* const* d_in, const int* in_sizes, int n_in,
                              void* d_out, int out_size, void* d_ws, size_t ws_size,
                              hipStream_t stream) {
    const float* x  = (const float*)d_in[0];
    const float* tg = (const float*)d_in[1];
    // d_in[2] = text_inputs (unused scalar)
    const float* Uw = (const float*)d_in[3];
    const float* Fw = (const float*)d_in[4];
    const float* Fb = (const float*)d_in[5];

    float* yout = (float*)d_out;
    float* loss = yout + (size_t)BB * YY;

    zero_loss_k<<<dim3(1), dim3(1), 0, stream>>>(loss);

    const dim3 grid((NTILES + WPB - 1) / WPB, BB);
    const bool pre = (d_ws != nullptr) && (ws_size >= PRE_BYTES);
    if (pre) {
        __bf16* xTg = (__bf16*)d_ws;
        xpose_k<<<dim3(NT16, BB), dim3(128), 0, stream>>>(x, xTg);
        laat_fused<true><<<grid, dim3(TPB), 0, stream>>>(
            x, xTg, tg, Uw, Fw, Fb, yout, loss);
    } else {
        laat_fused<false><<<grid, dim3(TPB), 0, stream>>>(
            x, nullptr, tg, Uw, Fw, Fb, yout, loss);
    }
}
